// SRNN_71382356459526
// MI455X (gfx1250) — matrix operand order repaired and checked by reading, not compiled, not run on hardware
//
#include <hip/hip_runtime.h>
#include <hip/hip_bf16.h>
#include <stdint.h>

// ---------------------------------------------------------------------------
// SRNN forward for MI455X (gfx1250, wave32, WMMA, TDM).
//
//  1) k_split:    W_h2h (f32) -> hi/lo bf16 pair (split precision).
//  2) k_gemm_i2h: XW[t*128+b][h] = x_t @ W_i2h^T + b_i2h + b_h2h, bf16 WMMA
//     16x16x32 with hi/lo split (3 products), f32 accumulate, LDS-staged
//     64x128 macro-tiles (full K-chunks use float4/global_load_b128).
//  3) k_recurrent: ONE persistent cooperative kernel, 72 workgroups, one
//     global atomic phase-barrier per timestep.  Per-step spike tiles are
//     staged global->LDS by the Tensor Data Mover (tensor_load_to_lds with
//     LDS padding so the DMA writes a bank-conflict-avoiding pitched layout),
//     tracked with s_wait_tensorcnt.
// ---------------------------------------------------------------------------

#ifndef USE_TDM
#define USE_TDM 1
#endif

typedef __attribute__((ext_vector_type(8)))  float          v8f;
typedef __attribute__((ext_vector_type(2)))  float          v2f;
typedef __attribute__((ext_vector_type(16))) __bf16         v16bf;
typedef __attribute__((ext_vector_type(8)))  unsigned short us8;
typedef __attribute__((ext_vector_type(16))) unsigned short us16;
typedef __attribute__((ext_vector_type(4)))  unsigned int   u32x4;
typedef __attribute__((ext_vector_type(8)))  int            i32x8;
typedef __attribute__((ext_vector_type(4)))  int            i32x4;

#define T_STEPS 500
#define BATCH   128
#define HID     1024
#define NIN     700
#define NOUT    20
#define NWG_H   64
#define NWG_O   8
#define NWG_TOT (NWG_H + NWG_O)

// LDS spike-tile pitch: 1024 data halves + 2x8 pad halves (TDM pad_amount=4
// DWORDs after every 256 DWORDs).  Pitch = 1040 halves = 2080 B = 520 DWORDs;
// 520 mod 64 = 8 -> at most 2-way bank conflict across the 16 fragment rows,
// and every +16B pad keeps ds_load_b128 chunks 16B-aligned.
#define APITCH 1040u
__device__ inline unsigned aidx(unsigned r, unsigned c) {
    return r * APITCH + c + ((c >= 512u) ? 8u : 0u);
}

// ---- small helpers --------------------------------------------------------

__device__ inline unsigned short f2bf(float f) {
    unsigned u = __float_as_uint(f);
    unsigned r = u + 0x7FFFu + ((u >> 16) & 1u);
    return (unsigned short)(r >> 16);
}
__device__ inline float bf2f(unsigned short h) {
    return __uint_as_float(((unsigned)h) << 16);
}

__device__ inline v16bf make_frag(const unsigned short* p0, const unsigned short* p1) {
    us8 a = *(const us8*)p0;
    us8 b = *(const us8*)p1;
    us16 v;
#pragma unroll
    for (int i = 0; i < 8; ++i) { v[i] = a[i]; v[8 + i] = b[i]; }
    return __builtin_bit_cast(v16bf, v);
}

__device__ inline v8f wmma_bf16(v16bf a, v16bf b, v8f c) {
    return __builtin_amdgcn_wmma_f32_16x16x32_bf16(false, a, false, b, (short)0, c, false, false);
}
__device__ inline v8f wmma_f32k4(v2f a, v2f b, v8f c) {
    return __builtin_amdgcn_wmma_f32_16x16x4_f32(false, a, false, b, (short)0, c, false, false);
}

// ---- TDM staging: 16x1024 bf16 tile, global -> padded LDS -----------------

__device__ inline void stage_spikes(const unsigned short* gsrc, unsigned short* lds) {
#if USE_TDM
    if ((threadIdx.x >> 5) == 0) {
        unsigned long long ga = (unsigned long long)(uintptr_t)gsrc;
        unsigned lds_off = (unsigned)(uintptr_t)lds;   // low 32 bits of generic ptr = LDS byte offset
        u32x4 g0;
        g0[0] = 1u;                                    // count=1, valid user descriptor
        g0[1] = lds_off;                               // lds_addr (bytes)
        g0[2] = (unsigned)(ga & 0xFFFFFFFFull);        // global_addr[31:0]
        g0[3] = (unsigned)((ga >> 32) & 0x1FFFFFFull)  // global_addr[56:32]
              | 0x80000000u;                           // type=2 ("image") in bits 127:126
        i32x8 g1;
        g1[0] = (int)(0x00010000u                      // data_size=1 -> 2 bytes
              | (1u << 20)                             // pad_enable
              | (7u << 22)                             // pad_interval: 256 DWORDs
              | (3u << 25));                           // pad_amount:   4 DWORDs
        g1[1] = (int)(1024u << 16);                    // tensor_dim0 = 1024 (low 16 in [63:48])
        g1[2] = (int)(16u << 16);                      // tensor_dim1 = 16   (low 16 in [95:80])
        g1[3] = (int)(1024u << 16);                    // tile_dim0 = 1024   ([127:112])
        g1[4] = 16;                                    // tile_dim1 = 16
        g1[5] = 1024;                                  // tensor_dim0_stride = 1024
        g1[6] = 0;
        g1[7] = 0;
        i32x4 z4; z4[0] = 0; z4[1] = 0; z4[2] = 0; z4[3] = 0;
        i32x8 z8;
#pragma unroll
        for (int i = 0; i < 8; ++i) z8[i] = 0;
        __builtin_amdgcn_tensor_load_to_lds(g0, g1, z4, z4, z8, 0);
        __builtin_amdgcn_s_wait_tensorcnt(0);
    }
#else
#pragma unroll
    for (int i = 0; i < 8; ++i) {
        unsigned u  = threadIdx.x + (unsigned)i * 256u;  // 2048 chunks of 8 halves
        unsigned r  = u >> 7;
        unsigned c8 = (u & 127u) * 8u;
        *(us8*)&lds[aidx(r, c8)] = *(const us8*)(gsrc + (size_t)r * HID + c8);
    }
#endif
    __syncthreads();
}

// ---- global phase barrier -------------------------------------------------

struct GBar { unsigned cnt; unsigned phase; };

__device__ inline void gbar_sync(GBar* b, unsigned target, unsigned nwg) {
    __threadfence();
    __syncthreads();
    if (threadIdx.x == 0) {
        unsigned prev = __hip_atomic_fetch_add(&b->cnt, 1u, __ATOMIC_ACQ_REL,
                                               __HIP_MEMORY_SCOPE_AGENT);
        if (prev == nwg - 1u) {
            __hip_atomic_store(&b->cnt, 0u, __ATOMIC_RELAXED, __HIP_MEMORY_SCOPE_AGENT);
            __hip_atomic_fetch_add(&b->phase, 1u, __ATOMIC_RELEASE, __HIP_MEMORY_SCOPE_AGENT);
        }
        while (__hip_atomic_load(&b->phase, __ATOMIC_ACQUIRE, __HIP_MEMORY_SCOPE_AGENT) < target) {
            __builtin_amdgcn_s_sleep(2);
        }
    }
    __syncthreads();
    __threadfence();
}

// ---- kernel 1: split W_h2h into hi/lo bf16 --------------------------------

__global__ void k_split(const float* __restrict__ W,
                        unsigned short* __restrict__ hi,
                        unsigned short* __restrict__ lo, int n) {
    int i = blockIdx.x * 256 + threadIdx.x;
    if (i < n) {
        float v = W[i];
        unsigned short h = f2bf(v);
        hi[i] = h;
        lo[i] = f2bf(v - bf2f(h));
    }
}

// ---- kernel 2: XW = x @ W_i2h^T + (b_i2h + b_h2h), split-bf16 WMMA --------

__global__ __launch_bounds__(256)
void k_gemm_i2h(const float* __restrict__ x,      // [128][500][700]
                const float* __restrict__ Wi,     // [1024][700]
                const float* __restrict__ b_i2h,  // [1024]
                const float* __restrict__ b_h2h,  // [1024]
                float* __restrict__ XW)           // [64000][1024]
{
    __shared__ __align__(16) unsigned short a_hi[64][40];
    __shared__ __align__(16) unsigned short a_lo[64][40];
    __shared__ __align__(16) unsigned short b_hi[128][40];
    __shared__ __align__(16) unsigned short b_lo[128][40];

    const unsigned tid  = threadIdx.x;
    const unsigned lane = tid & 31u;
    const unsigned w    = tid >> 5;
    const unsigned l16  = lane & 15u;
    const bool     hh   = lane >= 16u;
    const unsigned wm   = w >> 2;
    const unsigned wn   = w & 3u;
    const unsigned nt = blockIdx.x;       // 0..7
    const unsigned mt = blockIdx.y;       // 0..999
    const unsigned rows0 = mt * 64u;
    const unsigned cols0 = nt * 128u;

    v8f acc[2][2];
#pragma unroll
    for (int a = 0; a < 2; ++a)
#pragma unroll
        for (int b = 0; b < 2; ++b)
#pragma unroll
            for (int j = 0; j < 8; ++j) acc[a][b][j] = 0.0f;

    for (int k0 = 0; k0 < 704; k0 += 32) {
        const bool full = (k0 + 32 <= NIN);   // uniform: all but the last K-chunk
        // stage A (64 rows x 32 k)
#pragma unroll
        for (int i = 0; i < 2; ++i) {
            unsigned s = tid + i * 256u;
            unsigned r = s >> 3, kq = (s & 7u) * 4u;
            unsigned R = rows0 + r;
            unsigned bb = R & 127u, tt = R >> 7;
            const float* src = x + ((size_t)bb * T_STEPS + tt) * NIN + (k0 + kq);
            float v0, v1, v2, v3;
            if (full) {
                float4 f = *(const float4*)src;
                v0 = f.x; v1 = f.y; v2 = f.z; v3 = f.w;
            } else {
                int kb = k0 + (int)kq;
                v0 = (kb + 0 < NIN) ? src[0] : 0.0f;
                v1 = (kb + 1 < NIN) ? src[1] : 0.0f;
                v2 = (kb + 2 < NIN) ? src[2] : 0.0f;
                v3 = (kb + 3 < NIN) ? src[3] : 0.0f;
            }
            unsigned short h0 = f2bf(v0), h1 = f2bf(v1), h2 = f2bf(v2), h3 = f2bf(v3);
            a_hi[r][kq + 0] = h0; a_lo[r][kq + 0] = f2bf(v0 - bf2f(h0));
            a_hi[r][kq + 1] = h1; a_lo[r][kq + 1] = f2bf(v1 - bf2f(h1));
            a_hi[r][kq + 2] = h2; a_lo[r][kq + 2] = f2bf(v2 - bf2f(h2));
            a_hi[r][kq + 3] = h3; a_lo[r][kq + 3] = f2bf(v3 - bf2f(h3));
        }
        // stage B (128 rows x 32 k)
#pragma unroll
        for (int i = 0; i < 4; ++i) {
            unsigned s = tid + i * 256u;
            unsigned r = s >> 3, kq = (s & 7u) * 4u;
            unsigned n = cols0 + r;
            const float* src = Wi + (size_t)n * NIN + (k0 + kq);
            float v0, v1, v2, v3;
            if (full) {
                float4 f = *(const float4*)src;
                v0 = f.x; v1 = f.y; v2 = f.z; v3 = f.w;
            } else {
                int kb = k0 + (int)kq;
                v0 = (kb + 0 < NIN) ? src[0] : 0.0f;
                v1 = (kb + 1 < NIN) ? src[1] : 0.0f;
                v2 = (kb + 2 < NIN) ? src[2] : 0.0f;
                v3 = (kb + 3 < NIN) ? src[3] : 0.0f;
            }
            unsigned short h0 = f2bf(v0), h1 = f2bf(v1), h2 = f2bf(v2), h3 = f2bf(v3);
            b_hi[r][kq + 0] = h0; b_lo[r][kq + 0] = f2bf(v0 - bf2f(h0));
            b_hi[r][kq + 1] = h1; b_lo[r][kq + 1] = f2bf(v1 - bf2f(h1));
            b_hi[r][kq + 2] = h2; b_lo[r][kq + 2] = f2bf(v2 - bf2f(h2));
            b_hi[r][kq + 3] = h3; b_lo[r][kq + 3] = f2bf(v3 - bf2f(h3));
        }
        __syncthreads();

        const unsigned ao = hh ? 8u : 0u;
        const unsigned bo = hh ? 8u : 0u;
        v16bf afh[2], afl[2], bfh[2], bfl[2];
#pragma unroll
        for (int ms = 0; ms < 2; ++ms) {
            unsigned ar = wm * 32u + (unsigned)ms * 16u + l16;
            afh[ms] = make_frag(&a_hi[ar][ao], &a_hi[ar][16u + ao]);
            afl[ms] = make_frag(&a_lo[ar][ao], &a_lo[ar][16u + ao]);
        }
#pragma unroll
        for (int ns = 0; ns < 2; ++ns) {
            unsigned br = wn * 32u + (unsigned)ns * 16u + l16;
            bfh[ns] = make_frag(&b_hi[br][bo], &b_hi[br][bo + 16u]);
            bfl[ns] = make_frag(&b_lo[br][bo], &b_lo[br][bo + 16u]);
        }
#pragma unroll
        for (int ms = 0; ms < 2; ++ms)
#pragma unroll
            for (int ns = 0; ns < 2; ++ns) {
                acc[ms][ns] = wmma_bf16(afh[ms], bfh[ns], acc[ms][ns]); // hi*hi
                acc[ms][ns] = wmma_bf16(afh[ms], bfl[ns], acc[ms][ns]); // hi*lo
                acc[ms][ns] = wmma_bf16(afl[ms], bfh[ns], acc[ms][ns]); // lo*hi
            }
        __syncthreads();
    }

#pragma unroll
    for (int ms = 0; ms < 2; ++ms)
#pragma unroll
        for (int ns = 0; ns < 2; ++ns) {
            unsigned colg = cols0 + wn * 32u + (unsigned)ns * 16u + l16;
            float bias = b_i2h[colg] + b_h2h[colg];
#pragma unroll
            for (int j = 0; j < 8; ++j) {
                unsigned Rg = rows0 + wm * 32u + (unsigned)ms * 16u + (unsigned)j + (hh ? 8u : 0u);
                XW[(size_t)Rg * HID + colg] = acc[ms][ns][j] + bias;
            }
        }
}

// ---- kernel 3: persistent recurrent loop ----------------------------------

__global__ __launch_bounds__(256)
void k_recurrent(const float* __restrict__ XW,            // [500*128][1024]
                 const unsigned short* __restrict__ w_hi, // [1024][1024] bf16
                 const unsigned short* __restrict__ w_lo, // [1024][1024] bf16
                 const float* __restrict__ W_h2o,         // [20][1024]
                 const float* __restrict__ b_h2o,         // [20]
                 unsigned short* __restrict__ spk,        // [2][128][1024] bf16 (zeroed)
                 GBar* bar,
                 float* __restrict__ out)                 // [128][20][500]
{
    __shared__ __align__(16) unsigned short a_lds[16 * APITCH]; // padded spike tile
    __shared__ float partial[8][16][32];
    __shared__ float omem[16][20];

    const unsigned tid  = threadIdx.x;
    const unsigned lane = tid & 31u;
    const unsigned w    = tid >> 5;
    const unsigned l16  = lane & 15u;
    const bool     hh   = lane >= 16u;
    const unsigned g    = blockIdx.x;

    if (g < NWG_H) {
        // ---------------- hidden path ----------------
        const unsigned m   = g & 7u;
        const unsigned nb  = g >> 3;
        const unsigned col = nb * 128u + w * 16u + l16;
        const unsigned row0 = m * 16u;
        const unsigned short* wrow_hi = w_hi + (size_t)col * HID;
        const unsigned short* wrow_lo = w_lo + (size_t)col * HID;
        const unsigned ao = hh ? 8u : 0u;
        const unsigned bo = hh ? 8u : 0u;

        v8f mem;
#pragma unroll
        for (int j = 0; j < 8; ++j) mem[j] = 0.0f;

        for (int t = 0; t < T_STEPS; ++t) {
            const unsigned short* cur = spk + (size_t)(t & 1) * (BATCH * HID);
            unsigned short* nxt = spk + (size_t)((t + 1) & 1) * (BATCH * HID);

            stage_spikes(cur + (size_t)row0 * HID, a_lds);

            // acc starts as XW tile (includes both biases); prefetch next step
            v8f acc;
            const float* xwp = XW + ((size_t)t * BATCH + row0) * HID + col;
            __builtin_prefetch(xwp + (size_t)BATCH * HID, 0, 1);
#pragma unroll
            for (int j = 0; j < 8; ++j) {
                unsigned lm = (unsigned)j + (hh ? 8u : 0u);
                acc[j] = xwp[(size_t)lm * HID];
            }

            // acc += spk_prev @ (W_hi + W_lo)^T
#pragma unroll 4
            for (int kc = 0; kc < HID; kc += 32) {
                v16bf a  = make_frag(&a_lds[aidx(l16, kc + ao)],
                                     &a_lds[aidx(l16, kc + 16u + ao)]);
                v16bf bh = make_frag(wrow_hi + kc + bo, wrow_hi + kc + bo + 16);
                v16bf bl = make_frag(wrow_lo + kc + bo, wrow_lo + kc + bo + 16);
                acc = wmma_bf16(a, bh, acc);
                acc = wmma_bf16(a, bl, acc);
            }

            // leaky integrate + fire
#pragma unroll
            for (int j = 0; j < 8; ++j) {
                unsigned lm = (unsigned)j + (hh ? 8u : 0u);
                float mi = mem[j];
                float nm = (mi > 0.5f ? 0.0f : 0.5f * mi) + acc[j];
                mem[j] = nm;
                nxt[(size_t)(row0 + lm) * HID + col] =
                    (nm > 0.5f) ? (unsigned short)0x3F80u : (unsigned short)0u;
            }
            gbar_sync(bar, (unsigned)(t + 1), NWG_TOT);
        }
    } else {
        // ---------------- output path (exact f32 WMMA 16x16x4) ----------------
        const unsigned go   = g - NWG_H;
        const unsigned row0 = go * 16u;
        for (unsigned i = tid; i < 16u * 20u; i += 256u) ((float*)omem)[i] = 0.0f;
        __syncthreads();

        for (int t = 0; t < T_STEPS; ++t) {
            gbar_sync(bar, (unsigned)(t + 1), NWG_TOT);
            const unsigned short* sp = spk + (size_t)((t + 1) & 1) * (BATCH * HID);

            stage_spikes(sp + (size_t)row0 * HID, a_lds);

            v8f acc0, acc1;
#pragma unroll
            for (int j = 0; j < 8; ++j) { acc0[j] = 0.0f; acc1[j] = 0.0f; }
            const unsigned ko = hh ? 2u : 0u;
            const unsigned c0 = l16;
            const unsigned c1 = 16u + l16;
            const int kbeg = (int)(w * 128u), kend = kbeg + 128;
            for (int kc = kbeg; kc < kend; kc += 4) {
                v2f av;
                av.x = bf2f(a_lds[aidx(l16, (unsigned)kc + ko)]);
                av.y = bf2f(a_lds[aidx(l16, (unsigned)kc + ko + 1u)]);
                v2f b0;
                { const float* p = W_h2o + (size_t)c0 * HID + kc + ko; b0.x = p[0]; b0.y = p[1]; }
                v2f b1; b1.x = 0.0f; b1.y = 0.0f;
                if (c1 < NOUT) {
                    const float* p = W_h2o + (size_t)c1 * HID + kc + ko; b1.x = p[0]; b1.y = p[1];
                }
                acc0 = wmma_f32k4(av, b0, acc0);
                acc1 = wmma_f32k4(av, b1, acc1);
            }
#pragma unroll
            for (int j = 0; j < 8; ++j) {
                unsigned lm = (unsigned)j + (hh ? 8u : 0u);
                partial[w][lm][l16]       = acc0[j];
                partial[w][lm][16u + l16] = acc1[j];
            }
            __syncthreads();

            if (tid < 16u) {
                int r = (int)tid;
                float om[NOUT];
                float mx = -1e30f;
                for (int c = 0; c < NOUT; ++c) {
                    float s = b_h2o[c];
                    for (int ww = 0; ww < 8; ++ww) s += partial[ww][r][c];
                    float mi = omem[r][c];
                    float nm = (mi > 0.5f ? 0.0f : 0.5f * mi) + s;
                    omem[r][c] = nm;
                    om[c] = nm;
                    mx = fmaxf(mx, nm);
                }
                float se = 0.0f;
                for (int c = 0; c < NOUT; ++c) se += expf(om[c] - mx);
                float lse = mx + logf(se);
                size_t bb = (size_t)row0 + (size_t)r;
                for (int c = 0; c < NOUT; ++c)
                    out[(bb * NOUT + (size_t)c) * T_STEPS + t] = om[c] - lse;
            }
            __syncthreads();
        }
    }
}

// ---- host launcher --------------------------------------------------------

extern "C" void kernel_launch(void* const* d_in, const int* in_sizes, int n_in,
                              void* d_out, int out_size, void* d_ws, size_t ws_size,
                              hipStream_t stream) {
    (void)in_sizes; (void)n_in; (void)out_size; (void)ws_size;
    const float* x      = (const float*)d_in[0];
    const float* W_i2h  = (const float*)d_in[1];
    const float* b_i2h  = (const float*)d_in[2];
    const float* W_h2h  = (const float*)d_in[3];
    const float* b_h2h  = (const float*)d_in[4];
    const float* W_h2o  = (const float*)d_in[5];
    const float* b_h2o  = (const float*)d_in[6];
    float* out = (float*)d_out;

    char* ws = (char*)d_ws;
    float* XW = (float*)ws;
    size_t off = (size_t)T_STEPS * BATCH * HID * sizeof(float);       // 262.1 MB
    unsigned short* w_hi = (unsigned short*)(ws + off); off += (size_t)HID * HID * 2;
    unsigned short* w_lo = (unsigned short*)(ws + off); off += (size_t)HID * HID * 2;
    unsigned short* spkb = (unsigned short*)(ws + off);
    size_t spk_bytes = (size_t)2 * BATCH * HID * 2;
    off += spk_bytes;
    GBar* bar = (GBar*)(ws + off);

    (void)hipMemsetAsync(spkb, 0, spk_bytes + 256, stream);

    k_split<<<(HID * HID + 255) / 256, 256, 0, stream>>>(W_h2h, w_hi, w_lo, HID * HID);

    dim3 ggrid(8, 1000);
    k_gemm_i2h<<<ggrid, 256, 0, stream>>>(x, W_i2h, b_i2h, b_h2h, XW);

    k_recurrent<<<NWG_TOT, 256, 0, stream>>>(XW, w_hi, w_lo, W_h2o, b_h2o, spkb, bar, out);
}
